// MultiHeadSelfAttention_59571196395582
// MI455X (gfx1250) — compile-verified, matrix operand order checked by reading, not run on hardware
//
#include <hip/hip_runtime.h>
#include <hip/hip_bf16.h>

typedef __bf16 bf16_t;
typedef __attribute__((ext_vector_type(16))) __bf16 v16bf;
typedef __attribute__((ext_vector_type(8)))  __bf16 v8bf;
typedef __attribute__((ext_vector_type(8)))  float  v8f;

#define L_SEQ   3072
#define D_MODEL 1024
#define NHEAD   16
#define NKV     4
#define HDIM    64
#define W2      256          // window // 2
#define QB      (L_SEQ/16)   // 192 query blocks
#define WAVES   8

__device__ __forceinline__ v8f wmma_bf16(v16bf a, v16bf b, v8f c) {
  return __builtin_amdgcn_wmma_f32_16x16x32_bf16(false, a, false, b, (short)0, c,
                                                 false, false);
}

// Load a 16x32 bf16 fragment (A layout; B layout symmetric with N for M) from
// a row-major matrix, addressed as uniform base + 32-bit per-lane element
// offset (selects the saddr/voffset global-load form -> fewer address VGPRs).
// ISA 7.12.2: lanes 0-15 own K 0..7/16..23, lanes 16-31 own K 8..15/24..31.
__device__ __forceinline__ v16bf load_frag(const bf16_t* __restrict__ base, int off) {
  v8bf lo = *(const v8bf*)(base + off);        // K = +hh*8 .. +7
  v8bf hi = *(const v8bf*)(base + off + 16);   // K = +16+hh*8 .. +7
  v16bf r;
#pragma unroll
  for (int i = 0; i < 8; ++i) { r[i] = lo[i]; r[i + 8] = hi[i]; }
  return r;
}

// Per-lane base element offset for a fragment whose first row is `row`.
__device__ __forceinline__ int frag_off(int row, int ld, int k0, int lane) {
  int m = lane & 15, hh = lane >> 4;
  return (row + m) * ld + k0 + hh * 8;
}

// Register-blocked 32x64 output tile per wave: 2 A-frags x 4 B-frags
// -> 8 WMMAs per 6 fragment loads per K-step.
__device__ __forceinline__ void gemm_tile_32x64(const bf16_t* __restrict__ A, int lda, int m0,
                                                const bf16_t* __restrict__ B, int ldb, int n0,
                                                int K, int lane, v8f acc[2][4]) {
#pragma unroll
  for (int mi = 0; mi < 2; ++mi)
#pragma unroll
    for (int ni = 0; ni < 4; ++ni) acc[mi][ni] = (v8f){};
  int oa0 = frag_off(m0,      lda, 0, lane);
  int oa1 = frag_off(m0 + 16, lda, 0, lane);
  int ob0 = frag_off(n0,      ldb, 0, lane);
  int ob1 = frag_off(n0 + 16, ldb, 0, lane);
  int ob2 = frag_off(n0 + 32, ldb, 0, lane);
  int ob3 = frag_off(n0 + 48, ldb, 0, lane);
  for (int k0 = 0; k0 < K; k0 += 32) {
    if (k0 + 32 < K)  // weight-stream prefetch -> global_prefetch_b8
      __builtin_prefetch(B + ob0 + k0 + 32, 0, 1);
    v16bf a0 = load_frag(A, oa0 + k0);
    v16bf a1 = load_frag(A, oa1 + k0);
    v16bf b0 = load_frag(B, ob0 + k0);
    v16bf b1 = load_frag(B, ob1 + k0);
    v16bf b2 = load_frag(B, ob2 + k0);
    v16bf b3 = load_frag(B, ob3 + k0);
    acc[0][0] = wmma_bf16(a0, b0, acc[0][0]);
    acc[0][1] = wmma_bf16(a0, b1, acc[0][1]);
    acc[0][2] = wmma_bf16(a0, b2, acc[0][2]);
    acc[0][3] = wmma_bf16(a0, b3, acc[0][3]);
    acc[1][0] = wmma_bf16(a1, b0, acc[1][0]);
    acc[1][1] = wmma_bf16(a1, b1, acc[1][1]);
    acc[1][2] = wmma_bf16(a1, b2, acc[1][2]);
    acc[1][3] = wmma_bf16(a1, b3, acc[1][3]);
  }
}

__global__ void cvt_f32_bf16(const float* __restrict__ x, bf16_t* __restrict__ y, int n) {
  int i = blockIdx.x * blockDim.x + threadIdx.x;
  if (i < n) y[i] = (bf16_t)x[i];
}

// ---- QKV projection: qkv[L][1536] = x @ [Wq|Wk|Wv]^T + bias (bf16 out) ----
// 96 M-tiles x 24 N-tiles of 32x64. N-tile boundaries (1024, 1280) are
// multiples of 64, so each tile stays within one weight matrix.
__global__ void qkv_gemm(const bf16_t* __restrict__ xb,
                         const bf16_t* __restrict__ Wqb,
                         const bf16_t* __restrict__ Wkb,
                         const bf16_t* __restrict__ Wvb,
                         const float* __restrict__ bq,
                         const float* __restrict__ bk,
                         const float* __restrict__ bv,
                         bf16_t* __restrict__ qkv) {
  int lane = threadIdx.x & 31;
  int gw = blockIdx.x * (blockDim.x >> 5) + (threadIdx.x >> 5);
  int tm = gw / 24, tn = gw % 24;
  int m0 = tm * 32, n0 = tn * 64;
  const bf16_t* B; const float* bias; int nb;
  if (tn < 16)      { B = Wqb; bias = bq; nb = n0; }
  else if (tn < 20) { B = Wkb; bias = bk; nb = n0 - 1024; }
  else              { B = Wvb; bias = bv; nb = n0 - 1280; }
  v8f acc[2][4];
  gemm_tile_32x64(xb, D_MODEL, m0, B, D_MODEL, nb, D_MODEL, lane, acc);
  int n = lane & 15, hh = lane >> 4;
#pragma unroll
  for (int mi = 0; mi < 2; ++mi)
#pragma unroll
    for (int ni = 0; ni < 4; ++ni) {
      float bi = bias[nb + ni * 16 + n];
#pragma unroll
      for (int r = 0; r < 8; ++r) {
        int row = m0 + mi * 16 + r + 8 * hh;
        qkv[row * 1536 + n0 + ni * 16 + n] = (bf16_t)(acc[mi][ni][r] + bi);
      }
    }
}

// ---- RoPE on raw Q,K (pair d <-> d+32), writes Qr[L][1024], Kr[L][256] ----
__global__ void rope_kernel(const bf16_t* __restrict__ qkv,
                            bf16_t* __restrict__ Qr, bf16_t* __restrict__ Kr) {
  const int QP = L_SEQ * NHEAD * 32;            // query pairs
  int idx = blockIdx.x * blockDim.x + threadIdx.x;
  bool isQ = idx < QP;
  int t = isQ ? idx : idx - QP;
  int nh = isQ ? NHEAD : NKV;
  int d  = t % 32;
  int t2 = t / 32;
  int hd = t2 % nh;
  int l  = t2 / nh;
  int col = (isQ ? 0 : 1024) + hd * HDIM + d;
  float a = (float)qkv[l * 1536 + col];
  float b = (float)qkv[l * 1536 + col + 32];
  float invf = __expf(-((float)(2 * d) / (float)HDIM) * 9.2103403719761836f); // ln(1e4)
  float ang = (float)l * invf;
  float sn, cs;
  __sincosf(ang, &sn, &cs);
  float o0 = a * cs - b * sn;
  float o1 = b * cs + a * sn;
  bf16_t* dst = isQ ? Qr : Kr;
  int ldd = isQ ? (NHEAD * HDIM) : (NKV * HDIM);
  dst[l * ldd + hd * HDIM + d]      = (bf16_t)o0;
  dst[l * ldd + hd * HDIM + d + 32] = (bf16_t)o1;
}

// ---- sliding-window flash attention: one wave = (head, 16-query block) ----
__global__ void attn_kernel(const bf16_t* __restrict__ Qr,
                            const bf16_t* __restrict__ Kr,
                            const bf16_t* __restrict__ qkv,   // V at col 1280+
                            const int* __restrict__ amask,
                            bf16_t* __restrict__ ctx) {
  __shared__ __align__(16) bf16_t ldsV[WAVES][32 * HDIM]; // 32 keys x 64 dims
  __shared__ __align__(16) bf16_t ldsP[WAVES][16 * 32];   // 16 rows x 32 keys
  int lane = threadIdx.x & 31;
  int w = threadIdx.x >> 5;
  int gw = blockIdx.x * WAVES + w;
  int head = gw / QB;
  int qb = gw % QB;
  int q0 = qb * 16;
  int gk = head >> 2;                       // kv head (GQA group of 4)
  int n = lane & 15, hh = lane >> 4;

  // Q fragments are loop-invariant: 16x64 = two 16x32 A-frags.
  int oq = frag_off(q0, NHEAD * HDIM, head * HDIM, lane);
  v16bf aq0 = load_frag(Qr, oq);
  v16bf aq1 = load_frag(Qr, oq + 32);

  float mrow[8], lrow[8], alpha[8];
  v8f acc[4];
#pragma unroll
  for (int r = 0; r < 8; ++r) { mrow[r] = -1e30f; lrow[r] = 0.f; }
#pragma unroll
  for (int t = 0; t < 4; ++t) acc[t] = (v8f){};

  int cstart = q0 - W2; if (cstart < 0) cstart = 0; cstart &= ~31;
  int cend = q0 + 15 + W2; if (cend > L_SEQ - 1) cend = L_SEQ - 1;

  for (int kb = cstart; kb <= cend; kb += 32) {
    // Stage 32 V rows (this kv head) into per-wave LDS.
    {
      int voff = (kb + lane) * 1536 + 1280 + gk * HDIM;  // key < L (L%32==0)
      v8bf* dst = (v8bf*)&ldsV[w][lane * HDIM];
#pragma unroll
      for (int i = 0; i < 8; ++i) dst[i] = *(const v8bf*)(qkv + voff + i * 8);
    }
    // Scores: two 16-key subtiles, each = 2 WMMAs over HD=64.
    v8f s[2];
#pragma unroll
    for (int t = 0; t < 2; ++t) {
      int jb = kb + t * 16;
      int ok = frag_off(jb, NKV * HDIM, gk * HDIM, lane);
      v16bf bk0 = load_frag(Kr, ok);
      v16bf bk1 = load_frag(Kr, ok + 32);
      v8f sc = {};
      sc = wmma_bf16(aq0, bk0, sc);
      sc = wmma_bf16(aq1, bk1, sc);
      int j = jb + n;
      int am = amask[j];
#pragma unroll
      for (int r = 0; r < 8; ++r) {
        int i = q0 + r + 8 * hh;
        int dist = i - j; if (dist < 0) dist = -dist;
        float v = sc[r] * 0.125f;                 // 1/sqrt(64)
        if (dist > W2 || !am) v = -1e30f;
        sc[r] = v;
      }
      s[t] = sc;
    }
    // Streaming-softmax update (reductions stay inside 16-lane halves).
#pragma unroll
    for (int r = 0; r < 8; ++r) {
      float cm = fmaxf(s[0][r], s[1][r]);
      cm = fmaxf(cm, __shfl_xor(cm, 1, 32));
      cm = fmaxf(cm, __shfl_xor(cm, 2, 32));
      cm = fmaxf(cm, __shfl_xor(cm, 4, 32));
      cm = fmaxf(cm, __shfl_xor(cm, 8, 32));
      float mn = fmaxf(mrow[r], cm);
      float a0 = __expf(mrow[r] - mn);
      float p0 = (s[0][r] <= -1e29f) ? 0.f : __expf(s[0][r] - mn);
      float p1 = (s[1][r] <= -1e29f) ? 0.f : __expf(s[1][r] - mn);
      float rs = p0 + p1;
      rs += __shfl_xor(rs, 1, 32);
      rs += __shfl_xor(rs, 2, 32);
      rs += __shfl_xor(rs, 4, 32);
      rs += __shfl_xor(rs, 8, 32);
      lrow[r] = lrow[r] * a0 + rs;
      mrow[r] = mn;
      alpha[r] = a0;
      int M = r + 8 * hh;                         // C-layout row
      ldsP[w][M * 32 + n]      = (bf16_t)p0;      // re-layout C -> A via LDS
      ldsP[w][M * 32 + 16 + n] = (bf16_t)p1;
    }
#pragma unroll
    for (int t = 0; t < 4; ++t)
#pragma unroll
      for (int r = 0; r < 8; ++r) acc[t][r] *= alpha[r];

    // P as A-fragment (16x32), V as four B-fragments (32x16 each).
    v16bf pa = load_frag(&ldsP[w][0], frag_off(0, 32, 0, lane));
#pragma unroll
    for (int t = 0; t < 4; ++t) {
      v16bf bv;
#pragma unroll
      for (int v = 0; v < 8; ++v) {
        int g = v >> 2, j2 = v & 3;
        int k = g * 16 + hh * 8 + 2 * j2;
        int c = t * 16 + n;
        bv[2 * v]     = ldsV[w][k * HDIM + c];
        bv[2 * v + 1] = ldsV[w][(k + 1) * HDIM + c];
      }
      acc[t] = wmma_bf16(pa, bv, acc[t]);
    }
  }
  // Normalize and emit context (bf16) in [L][H*64].
#pragma unroll
  for (int r = 0; r < 8; ++r) {
    float inv = (lrow[r] > 0.f) ? (1.0f / lrow[r]) : 0.f;
    int row = q0 + r + 8 * hh;
#pragma unroll
    for (int t = 0; t < 4; ++t)
      ctx[row * D_MODEL + head * HDIM + t * 16 + n] = (bf16_t)(acc[t][r] * inv);
  }
}

// ---- output projection: out = ctx @ Wo^T + bo (fp32 out), 32x64 tiles ----
__global__ void oproj_gemm(const bf16_t* __restrict__ ctxb,
                           const bf16_t* __restrict__ Wob,
                           const float* __restrict__ bo,
                           float* __restrict__ out) {
  int lane = threadIdx.x & 31;
  int gw = blockIdx.x * (blockDim.x >> 5) + (threadIdx.x >> 5);
  int tm = gw >> 4, tn = gw & 15;               // 96 x 16 tiles of 32x64
  int m0 = tm * 32, n0 = tn * 64;
  v8f acc[2][4];
  gemm_tile_32x64(ctxb, D_MODEL, m0, Wob, D_MODEL, n0, D_MODEL, lane, acc);
  int n = lane & 15, hh = lane >> 4;
#pragma unroll
  for (int mi = 0; mi < 2; ++mi)
#pragma unroll
    for (int ni = 0; ni < 4; ++ni) {
      float bi = bo[n0 + ni * 16 + n];
#pragma unroll
      for (int r = 0; r < 8; ++r) {
        int row = m0 + mi * 16 + r + 8 * hh;
        out[row * D_MODEL + n0 + ni * 16 + n] = acc[mi][ni][r] + bi;
      }
    }
}

static inline size_t align256(size_t x) { return (x + 255) & ~(size_t)255; }

extern "C" void kernel_launch(void* const* d_in, const int* in_sizes, int n_in,
                              void* d_out, int out_size, void* d_ws, size_t ws_size,
                              hipStream_t stream) {
  const float* x  = (const float*)d_in[0];
  const int*   am = (const int*)d_in[1];
  const float* Wq = (const float*)d_in[2];
  const float* bq = (const float*)d_in[3];
  const float* Wk = (const float*)d_in[4];
  const float* bk = (const float*)d_in[5];
  const float* Wv = (const float*)d_in[6];
  const float* bv = (const float*)d_in[7];
  const float* Wo = (const float*)d_in[8];
  const float* bo = (const float*)d_in[9];
  float* out = (float*)d_out;

  char* ws = (char*)d_ws;
  size_t off = 0;
  bf16_t* xb  = (bf16_t*)(ws + off); off = align256(off + (size_t)L_SEQ * D_MODEL * 2);
  bf16_t* Wqb = (bf16_t*)(ws + off); off = align256(off + (size_t)D_MODEL * D_MODEL * 2);
  bf16_t* Wkb = (bf16_t*)(ws + off); off = align256(off + (size_t)256 * D_MODEL * 2);
  bf16_t* Wvb = (bf16_t*)(ws + off); off = align256(off + (size_t)256 * D_MODEL * 2);
  bf16_t* Wob = (bf16_t*)(ws + off); off = align256(off + (size_t)D_MODEL * D_MODEL * 2);
  bf16_t* qkv = (bf16_t*)(ws + off); off = align256(off + (size_t)L_SEQ * 1536 * 2);
  bf16_t* Qr  = (bf16_t*)(ws + off); off = align256(off + (size_t)L_SEQ * D_MODEL * 2);
  bf16_t* Kr  = (bf16_t*)(ws + off); off = align256(off + (size_t)L_SEQ * 256 * 2);
  bf16_t* ctx = (bf16_t*)(ws + off); off = align256(off + (size_t)L_SEQ * D_MODEL * 2);
  (void)ws_size; (void)n_in; (void)in_sizes; (void)out_size;

  const int T = 256;
  // 1) fp32 -> bf16 conversions
  int nx = L_SEQ * D_MODEL;
  cvt_f32_bf16<<<(nx + T - 1) / T, T, 0, stream>>>(x, xb, nx);
  int nw = D_MODEL * D_MODEL;
  cvt_f32_bf16<<<(nw + T - 1) / T, T, 0, stream>>>(Wq, Wqb, nw);
  int nk = 256 * D_MODEL;
  cvt_f32_bf16<<<(nk + T - 1) / T, T, 0, stream>>>(Wk, Wkb, nk);
  cvt_f32_bf16<<<(nk + T - 1) / T, T, 0, stream>>>(Wv, Wvb, nk);
  cvt_f32_bf16<<<(nw + T - 1) / T, T, 0, stream>>>(Wo, Wob, nw);

  // 2) fused QKV projection (WMMA, 32x64 tiles): 96*24 waves
  qkv_gemm<<<(96 * 24) / WAVES, T, 0, stream>>>(xb, Wqb, Wkb, Wvb,
                                                bq, bk, bv, qkv);
  // 3) RoPE
  int npair = L_SEQ * (NHEAD + NKV) * 32;
  rope_kernel<<<npair / T, T, 0, stream>>>(qkv, Qr, Kr);

  // 4) sliding-window flash attention (WMMA)
  attn_kernel<<<(NHEAD * QB) / WAVES, T, 0, stream>>>(Qr, Kr, qkv, am, ctx);

  // 5) output projection (WMMA, 32x64 tiles): 96*16 waves
  oproj_gemm<<<(96 * 16) / WAVES, T, 0, stream>>>(ctx, Wob, bo, out);
}
